// TinyViT_32306744000560
// MI455X (gfx1250) — matrix operand order repaired and checked by reading, not compiled, not run on hardware
//
#include <hip/hip_runtime.h>
#include <hip/hip_bf16.h>

typedef __attribute__((ext_vector_type(16))) _Float16 v16h;
typedef __attribute__((ext_vector_type(8)))  float    v8f;
typedef unsigned int u32;
typedef unsigned long long u64;

#define DIM   160
#define HEADS 5
#define KD    32
#define HID   640
#define NWIN  400        // 16 batches * 5*5 windows
#define NTOK  224        // padded window tokens (196 -> 224)
#define NVAL  196
#define ROWS_ATTN (NWIN*NTOK)   // 89600 = 700*128
#define ROWS_MLP  (16*4096)     // 65536 = 512*128
#define NPIX  (16*4096*160)     // elements in residual stream

union AFu { v16h v; u32 u[8]; _Float16 h[16]; };

__device__ __forceinline__ v8f wmma_f16(v16h a, v16h b, v8f c) {
  return __builtin_amdgcn_wmma_f32_16x16x32_f16(false, a, false, b, (short)0, c, false, false);
}

// async copy 16B global -> LDS (per-lane addresses), ASYNCcnt-tracked
__device__ __forceinline__ void async_b128(u32 lds_off, const void* gptr) {
  asm volatile("global_load_async_to_lds_b128 %0, %1, off"
               :: "v"(lds_off), "v"((u64)gptr) : "memory");
}
__device__ __forceinline__ void async_b128_o(u32 lds_off, const void* gptr, int imm_off) {
  // INST_OFFSET is added to BOTH the LDS and the global address (ISA 15.18.3)
  if (imm_off == 16)
    asm volatile("global_load_async_to_lds_b128 %0, %1, off offset:16"
                 :: "v"(lds_off), "v"((u64)gptr) : "memory");
  else if (imm_off == 32)
    asm volatile("global_load_async_to_lds_b128 %0, %1, off offset:32"
                 :: "v"(lds_off), "v"((u64)gptr) : "memory");
  else
    asm volatile("global_load_async_to_lds_b128 %0, %1, off offset:48"
                 :: "v"(lds_off), "v"((u64)gptr) : "memory");
}
__device__ __forceinline__ void wait_async0() {
  asm volatile("s_wait_asynccnt 0x0" ::: "memory");
}
__device__ __forceinline__ u32 lds_addr(const void* p) {
  return (u32)(size_t)p;   // generic LDS pointer: low 32 bits = LDS offset
}

// ---------------------------------------------------------------------------
// f32 -> f16 convert (weights)
// ---------------------------------------------------------------------------
__global__ void cvt_f16_kernel(const float* __restrict__ in, _Float16* __restrict__ out, int n) {
  for (int i = blockIdx.x * 256 + threadIdx.x; i < n; i += gridDim.x * 256)
    out[i] = (_Float16)in[i];
}

// ---------------------------------------------------------------------------
// LN1 + window partition/pad: x[16,4096,160] -> xn16[400,224,160] f16
// ---------------------------------------------------------------------------
__global__ __launch_bounds__(256) void ln1_window_kernel(
    const float* __restrict__ x, const float* __restrict__ g,
    const float* __restrict__ b, _Float16* __restrict__ out) {
  int gidx = blockIdx.x * 8 + (threadIdx.x >> 5);       // row over 400*224
  int lane = threadIdx.x & 31;
  int w = gidx / NTOK, t = gidx % NTOK;
  size_t obase = (size_t)gidx * DIM;
  if (t >= NVAL) {
    #pragma unroll
    for (int j = 0; j < 5; ++j) out[obase + lane + 32 * j] = (_Float16)0.0f;
    return;
  }
  int bb = w / 25, wrem = w % 25, wh = wrem / 5, ww = wrem % 5;
  int ti = t / 14, tj = t % 14;
  int gh = wh * 14 + ti, gw = ww * 14 + tj;
  bool valid = (gh < 64) && (gw < 64);
  const float* xp = x + ((size_t)bb * 4096 + gh * 64 + gw) * DIM;
  float v[5];
  #pragma unroll
  for (int j = 0; j < 5; ++j) v[j] = valid ? xp[lane + 32 * j] : 0.0f;
  float s = v[0] + v[1] + v[2] + v[3] + v[4];
  #pragma unroll
  for (int off = 16; off >= 1; off >>= 1) s += __shfl_xor(s, off, 32);
  float mean = s * (1.0f / DIM);
  float vs = 0.f;
  #pragma unroll
  for (int j = 0; j < 5; ++j) { float d = v[j] - mean; vs += d * d; }
  #pragma unroll
  for (int off = 16; off >= 1; off >>= 1) vs += __shfl_xor(vs, off, 32);
  float rs = rsqrtf(vs * (1.0f / DIM) + 1e-5f);
  #pragma unroll
  for (int j = 0; j < 5; ++j) {
    int c = lane + 32 * j;
    out[obase + c] = (_Float16)((v[j] - mean) * rs * g[c] + b[c]);
  }
}

// ---------------------------------------------------------------------------
// LN2: x2[65536,160] f32 -> xn2[65536,160] f16
// ---------------------------------------------------------------------------
__global__ __launch_bounds__(256) void ln2_kernel(
    const float* __restrict__ x, const float* __restrict__ g,
    const float* __restrict__ b, _Float16* __restrict__ out) {
  int gidx = blockIdx.x * 8 + (threadIdx.x >> 5);
  int lane = threadIdx.x & 31;
  const float* xp = x + (size_t)gidx * DIM;
  float v[5];
  #pragma unroll
  for (int j = 0; j < 5; ++j) v[j] = xp[lane + 32 * j];
  float s = v[0] + v[1] + v[2] + v[3] + v[4];
  #pragma unroll
  for (int off = 16; off >= 1; off >>= 1) s += __shfl_xor(s, off, 32);
  float mean = s * (1.0f / DIM);
  float vs = 0.f;
  #pragma unroll
  for (int j = 0; j < 5; ++j) { float d = v[j] - mean; vs += d * d; }
  #pragma unroll
  for (int off = 16; off >= 1; off >>= 1) vs += __shfl_xor(vs, off, 32);
  float rs = rsqrtf(vs * (1.0f / DIM) + 1e-5f);
  #pragma unroll
  for (int j = 0; j < 5; ++j) {
    int c = lane + 32 * j;
    out[(size_t)gidx * DIM + c] = (_Float16)((v[j] - mean) * rs * g[c] + b[c]);
  }
}

// ---------------------------------------------------------------------------
// Generic WMMA GEMM: C[M,N] = A[M,K] @ B[N,K]^T + bias, f16 operands f32 acc.
// Tiles staged with GLOBAL_LOAD_ASYNC_TO_LDS_B128 (ASYNCcnt).
// block 256 threads = 8 waves, tile 128(M) x 64(N), each wave 32x32.
// EPI: 0 bias->f16 | 1 bias+GELU->f16 | 2 bias+res->f32 | 3 bias->f32
// ---------------------------------------------------------------------------
template<int EPI>
__global__ __launch_bounds__(256) void gemm_f16_kernel(
    const _Float16* __restrict__ A, const _Float16* __restrict__ B,
    const float* __restrict__ bias, void* __restrict__ C,
    const float* __restrict__ res, int M, int N, int K) {
  __shared__ _Float16 As[128 * 40];   // 32 K + 8 pad
  __shared__ _Float16 Bs[64 * 40];
  const int tid = threadIdx.x;
  const int lane = tid & 31, wave = tid >> 5;
  const int l16 = lane & 15, laneHi = lane >> 4;
  const int m0 = blockIdx.x * 128, n0 = blockIdx.y * 64;
  const int wm = (wave >> 1) * 32, wn = (wave & 1) * 32;

  const int rA = tid >> 1, ccA = (tid & 1) * 16;   // A: 128 rows x 2 chunks(16 f16)
  const int rB = tid >> 2, ccB = (tid & 3) * 8;    // B: 64 rows x 4 chunks(8 f16)
  const bool aOk = (m0 + rA) < M;
  const bool bOk = (n0 + rB) < N;
  // OOB rows: zero once; predicated async loads never touch them again
  if (!aOk) {
    *(uint4*)(&As[rA * 40 + ccA])     = (uint4){0,0,0,0};
    *(uint4*)(&As[rA * 40 + ccA + 8]) = (uint4){0,0,0,0};
  }
  if (!bOk) *(uint4*)(&Bs[rB * 40 + ccB]) = (uint4){0,0,0,0};

  v8f acc[2][2];
  #pragma unroll
  for (int i = 0; i < 2; ++i)
    #pragma unroll
    for (int j = 0; j < 2; ++j)
      acc[i][j] = (v8f){0.f,0.f,0.f,0.f,0.f,0.f,0.f,0.f};

  const u32 ldsA = lds_addr(&As[rA * 40 + ccA]);
  const u32 ldsB = lds_addr(&Bs[rB * 40 + ccB]);

  for (int k0 = 0; k0 < K; k0 += 32) {
    if (aOk) {
      const _Float16* src = A + (size_t)(m0 + rA) * K + k0 + ccA;
      async_b128(ldsA, src);
      async_b128_o(ldsA, src, 16);    // second 16B: same +16 delta on LDS & global
    }
    if (bOk) async_b128(ldsB, B + (size_t)(n0 + rB) * K + k0 + ccB);
    wait_async0();
    __syncthreads();

    AFu af[2], bf[2];
    #pragma unroll
    for (int fm = 0; fm < 2; ++fm) {
      const u32* rp = (const u32*)(&As[(wm + fm * 16 + l16) * 40]);
      #pragma unroll
      for (int j = 0; j < 8; ++j) af[fm].u[j] = rp[(j < 4 ? j : j + 4) + laneHi * 4];
    }
    #pragma unroll
    for (int fn = 0; fn < 2; ++fn) {
      const u32* bp = (const u32*)(&Bs[(wn + fn * 16 + l16) * 40]);
      #pragma unroll
      for (int j = 0; j < 8; ++j) bf[fn].u[j] = bp[(j < 4 ? j : j + 4) + laneHi * 4];
    }
    #pragma unroll
    for (int fm = 0; fm < 2; ++fm)
      #pragma unroll
      for (int fn = 0; fn < 2; ++fn)
        acc[fm][fn] = wmma_f16(af[fm].v, bf[fn].v, acc[fm][fn]);
    __syncthreads();
  }

  #pragma unroll
  for (int fm = 0; fm < 2; ++fm)
    #pragma unroll
    for (int fn = 0; fn < 2; ++fn) {
      int col = n0 + wn + fn * 16 + l16;
      float bv = (col < N) ? bias[col] : 0.f;
      #pragma unroll
      for (int i = 0; i < 8; ++i) {
        int row = m0 + wm + fm * 16 + i + laneHi * 8;
        if (row < M && col < N) {
          float v = acc[fm][fn][i] + bv;
          if (EPI == 1) v = 0.5f * v * (1.0f + erff(v * 0.70710678118f));
          if (EPI == 2) v += res[(size_t)row * N + col];
          if (EPI == 0 || EPI == 1)
            ((_Float16*)C)[(size_t)row * N + col] = (_Float16)v;
          else
            ((float*)C)[(size_t)row * N + col] = v;
        }
      }
    }
}

// ---------------------------------------------------------------------------
// Flash-style windowed attention. qkv16[89600,480] f16; per row, head h:
// cols [h*96..+31]=Q, +32..63=K, +64..95=V.
// One block per (window, head): 7 waves x 2 row tiles. K staged to LDS via
// async b128; V staged transposed (d-major) so AV B-fragments are dword loads.
// ---------------------------------------------------------------------------
__global__ __launch_bounds__(224) void attn_kernel(
    const _Float16* __restrict__ qkv, const float* __restrict__ biases,
    const int* __restrict__ bidx, _Float16* __restrict__ out) {
  __shared__ _Float16 Ks[NTOK * 40];     // 224 rows x (32+8)
  __shared__ _Float16 VsT[32 * 232];     // 32 d-rows x (224+8) tokens
  __shared__ _Float16 pbuf[7][16 * 40];
  const int wid = blockIdx.x;
  const int w = wid / HEADS, h = wid % HEADS;
  const int tid = threadIdx.x;
  const int lane = tid & 31, wave = tid >> 5;
  const int l16 = lane & 15, laneHi = lane >> 4;
  const size_t rowbase = (size_t)w * NTOK;
  const _Float16* qp = qkv + rowbase * 480 + h * 96;
  const float scale = 0.17677669529f;    // 1/sqrt(32)

  { // stage K row tid (32 f16 = 64B) via async copies: LDS/global deltas match
    const _Float16* ksrc = qp + 32 + (size_t)tid * 480;
    u32 kl = lds_addr(&Ks[tid * 40]);
    async_b128(kl, ksrc);
    async_b128_o(kl, ksrc, 16);
    async_b128_o(kl, ksrc, 32);
    async_b128_o(kl, ksrc, 48);
  }
  { // stage V row tid transposed: VsT[d][tok]
    union { uint4 q[4]; _Float16 hh[32]; } vrow;
    const uint4* vsrc = (const uint4*)(qp + 64 + (size_t)tid * 480);
    #pragma unroll
    for (int j = 0; j < 4; ++j) vrow.q[j] = vsrc[j];
    #pragma unroll
    for (int d = 0; d < 32; ++d) VsT[d * 232 + tid] = vrow.hh[d];
  }
  wait_async0();
  __syncthreads();

  for (int rt = wave; rt < 14; rt += 7) {
    AFu qa;
    {
      const u32* qr = (const u32*)(qp + (size_t)(rt * 16 + l16) * 480);
      #pragma unroll
      for (int j = 0; j < 8; ++j) qa.u[j] = qr[(j < 4 ? j : j + 4) + laneHi * 4];
    }
    float Mrow[8], Srow[8];
    #pragma unroll
    for (int i = 0; i < 8; ++i) { Mrow[i] = -1e30f; Srow[i] = 0.f; }
    v8f accd0 = (v8f){0.f,0.f,0.f,0.f,0.f,0.f,0.f,0.f};
    v8f accd1 = (v8f){0.f,0.f,0.f,0.f,0.f,0.f,0.f,0.f};

    for (int ct2 = 0; ct2 < 7; ++ct2) {
      float sv[2][8];
      #pragma unroll
      for (int half2 = 0; half2 < 2; ++half2) {
        int ct = ct2 * 2 + half2;
        AFu kb;
        const u32* kr = (const u32*)(&Ks[(ct * 16 + l16) * 40]);
        #pragma unroll
        for (int j = 0; j < 8; ++j) kb.u[j] = kr[(j < 4 ? j : j + 4) + laneHi * 4];
        v8f z = (v8f){0.f,0.f,0.f,0.f,0.f,0.f,0.f,0.f};
        v8f s = wmma_f16(qa.v, kb.v, z);
        #pragma unroll
        for (int i = 0; i < 8; ++i) {
          int r = rt * 16 + i + laneHi * 8;
          int c = ct * 16 + l16;
          float val = -1e30f;
          if (c < NVAL) {
            int rb = (r < NVAL) ? r : 0;
            val = s[i] * scale + biases[h * 196 + bidx[rb * 196 + c]];
          }
          sv[half2][i] = val;
        }
      }
      float p0[8], p1[8];
      #pragma unroll
      for (int i = 0; i < 8; ++i) {
        float m2 = fmaxf(sv[0][i], sv[1][i]);
        #pragma unroll
        for (int off = 8; off >= 1; off >>= 1) m2 = fmaxf(m2, __shfl_xor(m2, off, 16));
        float newM = fmaxf(Mrow[i], m2);
        p0[i] = __expf(sv[0][i] - newM);
        p1[i] = __expf(sv[1][i] - newM);
        float ts = p0[i] + p1[i];
        #pragma unroll
        for (int off = 8; off >= 1; off >>= 1) ts += __shfl_xor(ts, off, 16);
        float f = __expf(Mrow[i] - newM);
        Srow[i] = Srow[i] * f + ts;
        Mrow[i] = newM;
        accd0[i] *= f;
        accd1[i] *= f;
      }
      // transpose P (C layout -> A layout) through wave-private LDS
      _Float16* pb = &pbuf[wave][0];
      #pragma unroll
      for (int i = 0; i < 8; ++i) {
        pb[(i + laneHi * 8) * 40 + l16]      = (_Float16)p0[i];
        pb[(i + laneHi * 8) * 40 + 16 + l16] = (_Float16)p1[i];
      }
      asm volatile("s_wait_dscnt 0x0" ::: "memory");
      AFu pa;
      {
        const u32* pr = (const u32*)(pb + l16 * 40);
        #pragma unroll
        for (int j = 0; j < 8; ++j) pa.u[j] = pr[(j < 4 ? j : j + 4) + laneHi * 4];
      }
      // V fragments from transposed LDS: consecutive-token dwords per d-row
      AFu vb0, vb1;
      {
        const u32* v0p = (const u32*)(&VsT[l16 * 232 + ct2 * 32]);
        const u32* v1p = (const u32*)(&VsT[(16 + l16) * 232 + ct2 * 32]);
        #pragma unroll
        for (int j = 0; j < 8; ++j) {
          vb0.u[j] = v0p[(j < 4 ? j : j + 4) + laneHi * 4];
          vb1.u[j] = v1p[(j < 4 ? j : j + 4) + laneHi * 4];
        }
      }
      accd0 = wmma_f16(pa.v, vb0.v, accd0);
      accd1 = wmma_f16(pa.v, vb1.v, accd1);
    }
    #pragma unroll
    for (int i = 0; i < 8; ++i) {
      int r = rt * 16 + i + laneHi * 8;
      float inv = __builtin_amdgcn_rcpf(Srow[i]);
      out[(rowbase + r) * DIM + h * KD + l16]      = (_Float16)(accd0[i] * inv);
      out[(rowbase + r) * DIM + h * KD + 16 + l16] = (_Float16)(accd1[i] * inv);
    }
  }
}

// ---------------------------------------------------------------------------
// Window un-partition + residual: x1 = x + scatter(proj)
// ---------------------------------------------------------------------------
__global__ __launch_bounds__(256) void attn_residual_kernel(
    const float* __restrict__ x, const float* __restrict__ proj, float* __restrict__ y) {
  int idx = blockIdx.x * 256 + threadIdx.x;
  if (idx >= NPIX) return;
  int c = idx % DIM, p = idx / DIM;
  int gw = p % 64, gh = (p / 64) % 64, bb = p / 4096;
  int wh = gh / 14, ti = gh % 14, ww = gw / 14, tj = gw % 14;
  int w = (bb * 5 + wh) * 5 + ww;
  int t = ti * 14 + tj;
  y[idx] = x[idx] + proj[((size_t)w * NTOK + t) * DIM + c];
}

// ---------------------------------------------------------------------------
// Depthwise 3x3 conv (NHWC) + BN affine
// ---------------------------------------------------------------------------
__global__ __launch_bounds__(256) void dwconv_bn_kernel(
    const float* __restrict__ x, const float* __restrict__ cw,
    const float* __restrict__ bg, const float* __restrict__ bb2,
    float* __restrict__ y) {
  int idx = blockIdx.x * 256 + threadIdx.x;
  if (idx >= NPIX) return;
  int c = idx % DIM, p = idx / DIM;
  int gw = p % 64, gh = (p / 64) % 64, bbi = p / 4096;
  float acc = 0.f;
  #pragma unroll
  for (int dh = 0; dh < 3; ++dh)
    #pragma unroll
    for (int dw = 0; dw < 3; ++dw) {
      int hh = gh + dh - 1, wp = gw + dw - 1;
      if (hh >= 0 && hh < 64 && wp >= 0 && wp < 64)
        acc += x[((size_t)bbi * 4096 + hh * 64 + wp) * DIM + c] * cw[c * 9 + dh * 3 + dw];
    }
  y[idx] = acc * (bg[c] * rsqrtf(1.0f + 1e-5f)) + bb2[c];
}

// ---------------------------------------------------------------------------
extern "C" void kernel_launch(void* const* d_in, const int* in_sizes, int n_in,
                              void* d_out, int out_size, void* d_ws, size_t ws_size,
                              hipStream_t stream) {
  const float* x        = (const float*)d_in[0];
  const float* ln1_g    = (const float*)d_in[1];
  const float* ln1_b    = (const float*)d_in[2];
  const float* qkv_w    = (const float*)d_in[3];
  const float* qkv_b    = (const float*)d_in[4];
  const float* proj_w   = (const float*)d_in[5];
  const float* proj_b   = (const float*)d_in[6];
  const float* attn_bi  = (const float*)d_in[7];
  const float* conv_w   = (const float*)d_in[8];
  const float* bn_g     = (const float*)d_in[9];
  const float* bn_b     = (const float*)d_in[10];
  const float* ln2_g    = (const float*)d_in[11];
  const float* ln2_b    = (const float*)d_in[12];
  const float* fc1_w    = (const float*)d_in[13];
  const float* fc1_b    = (const float*)d_in[14];
  const float* fc2_w    = (const float*)d_in[15];
  const float* fc2_b    = (const float*)d_in[16];
  const int*   bias_idx = (const int*)d_in[17];
  float* out = (float*)d_out;

  char* ws = (char*)d_ws;
  size_t off = 0;
  auto take = [&](size_t bytes) { char* p = ws + off; off += (bytes + 255) & ~(size_t)255; return p; };
  _Float16* wq16   = (_Float16*)take((size_t)480 * 160 * 2);
  _Float16* wp16   = (_Float16*)take((size_t)160 * 160 * 2);
  _Float16* wf1_16 = (_Float16*)take((size_t)640 * 160 * 2);
  _Float16* wf2_16 = (_Float16*)take((size_t)160 * 640 * 2);
  _Float16* xn16   = (_Float16*)take((size_t)ROWS_ATTN * DIM * 2);   // reused as xn2
  _Float16* qkv16  = (_Float16*)take((size_t)ROWS_ATTN * 480 * 2);   // reused as h16
  _Float16* attn16 = (_Float16*)take((size_t)ROWS_ATTN * DIM * 2);
  float*    projf  = (float*)   take((size_t)ROWS_ATTN * DIM * 4);
  float*    x1     = (float*)   take((size_t)NPIX * 4);
  float*    x2     = (float*)   take((size_t)NPIX * 4);
  _Float16* xn2    = xn16;
  _Float16* h16    = qkv16;

  // 0) weights -> f16
  cvt_f16_kernel<<<300, 256, 0, stream>>>(qkv_w,  wq16,   480 * 160);
  cvt_f16_kernel<<<100, 256, 0, stream>>>(proj_w, wp16,   160 * 160);
  cvt_f16_kernel<<<400, 256, 0, stream>>>(fc1_w,  wf1_16, 640 * 160);
  cvt_f16_kernel<<<400, 256, 0, stream>>>(fc2_w,  wf2_16, 160 * 640);

  // 1) LN1 + window partition
  ln1_window_kernel<<<ROWS_ATTN / 8, 256, 0, stream>>>(x, ln1_g, ln1_b, xn16);

  // 2) QKV GEMM: [89600,160] @ [480,160]^T -> f16
  gemm_f16_kernel<0><<<dim3(ROWS_ATTN / 128, (480 + 63) / 64), 256, 0, stream>>>(
      xn16, wq16, qkv_b, qkv16, nullptr, ROWS_ATTN, 480, 160);

  // 3) windowed flash attention
  attn_kernel<<<NWIN * HEADS, 224, 0, stream>>>(qkv16, attn_bi, bias_idx, attn16);

  // 4) proj GEMM -> f32
  gemm_f16_kernel<3><<<dim3(ROWS_ATTN / 128, (DIM + 63) / 64), 256, 0, stream>>>(
      attn16, wp16, proj_b, projf, nullptr, ROWS_ATTN, DIM, 160);

  // 5) un-window + residual
  attn_residual_kernel<<<NPIX / 256, 256, 0, stream>>>(x, projf, x1);

  // 6) depthwise conv + BN (replaces residual stream)
  dwconv_bn_kernel<<<NPIX / 256, 256, 0, stream>>>(x1, conv_w, bn_g, bn_b, x2);

  // 7) LN2
  ln2_kernel<<<ROWS_MLP / 8, 256, 0, stream>>>(x2, ln2_g, ln2_b, xn2);

  // 8) fc1 + exact GELU -> f16
  gemm_f16_kernel<1><<<dim3(ROWS_MLP / 128, HID / 64), 256, 0, stream>>>(
      xn2, wf1_16, fc1_b, h16, nullptr, ROWS_MLP, HID, 160);

  // 9) fc2 + residual -> out f32
  gemm_f16_kernel<2><<<dim3(ROWS_MLP / 128, (DIM + 63) / 64), 256, 0, stream>>>(
      h16, wf2_16, fc2_b, out, x2, ROWS_MLP, DIM, 640);
}
